// MultilayerGRU_60833916780788
// MI455X (gfx1250) — matrix operand order repaired and checked by reading, not compiled, not run on hardware
//
#include <hip/hip_runtime.h>
#include <hip/hip_bf16.h>
#include <stdint.h>

// ---------------------------------------------------------------------------
// MultilayerGRU for MI455X (gfx1250): wave32 + WMMA f16 (16x16x32), TDM
// (tensor_load_to_lds) staging of LDS-resident recurrent weights + hidden
// state, persistent 8-WG scan with grid barrier.
// ---------------------------------------------------------------------------

typedef _Float16 h8v  __attribute__((ext_vector_type(8)));
typedef _Float16 h16v __attribute__((ext_vector_type(16)));
typedef float    f8v  __attribute__((ext_vector_type(8)));
typedef unsigned int u32x4 __attribute__((ext_vector_type(4)));
typedef int          i32x4 __attribute__((ext_vector_type(4)));
typedef int          i32x8 __attribute__((ext_vector_type(8)));

namespace gru5 {
constexpr int B   = 32;
constexpr int S   = 512;
constexpr int DI  = 256;
constexpr int DH  = 512;
constexpr int DO  = 128;
constexpr int M   = B * S;     // 16384 rows, row index = s*B + b
constexpr int G3  = 3 * DH;    // 1536 (z|r|n)
constexpr int NWG = 8;         // persistent workgroups in the scan
constexpr unsigned SLAB_BYTES = 8u * 16 * DH * sizeof(_Float16);  // 128 KB per slab
}

// ---- WMMA fragment helpers (layouts per CDNA5 ISA 7.12.2) ------------------
__device__ __forceinline__ h16v gru5_cat16(h8v lo, h8v hi) {
  return __builtin_shufflevector(lo, hi, 0,1,2,3,4,5,6,7,8,9,10,11,12,13,14,15);
}
// A (16x32 f16, row-major source): lane<16 -> row m0+lane, K {k0..k0+7, k0+16..k0+23}
__device__ __forceinline__ h16v gru5_afrag(const _Float16* base, int ld, int m0, int k0, int lane) {
  const _Float16* p = base + (size_t)(m0 + (lane & 15)) * ld + k0 + ((lane >> 4) << 3);
  return gru5_cat16(*(const h8v*)p, *(const h8v*)(p + 16));
}
// B (32x16 f16) from a pre-transposed [N][K] weight image.
__device__ __forceinline__ h16v gru5_bfrag(const _Float16* bt, int ld, int n0, int k0, int lane) {
  const _Float16* p = bt + (size_t)(n0 + (lane & 15)) * ld + k0 + ((lane >> 4) << 3);
  return gru5_cat16(*(const h8v*)p, *(const h8v*)(p + 16));
}
__device__ __forceinline__ f8v gru5_wmma(h16v a, h16v b, f8v c) {
  return __builtin_amdgcn_wmma_f32_16x16x32_f16(false, a, false, b, (short)0, c, false, false);
}
__device__ __forceinline__ float gru5_sigmoid(float x) { return 1.0f / (1.0f + __expf(-x)); }

// ---- TDM: 1-D contiguous global -> LDS DMA (CDNA5 ISA 08, D# groups 0/1) ---
// data_size_code: 0=1B 1=2B 2=4B 3=8B. elems = elements of that size (<=65535).
// 6-arg builtin form (this toolchain): (g0, g1, g2, g3, g4, cpol).
__device__ __forceinline__ void gru5_tdm_load_1d(const void* gsrc, unsigned lds_off,
                                                 unsigned elems, unsigned data_size_code) {
  const unsigned long long ga = (unsigned long long)gsrc;
  u32x4 g0;
  g0[0] = 1u;                                                // count=1, user mode
  g0[1] = lds_off;                                           // lds_addr (bytes)
  g0[2] = (unsigned)(ga & 0xffffffffu);                      // global_addr[31:0]
  g0[3] = (unsigned)((ga >> 32) & 0x01ffffffu) | (2u << 30); // global_addr[56:32] | type=2
  i32x8 g1;
  g1[0] = (int)(data_size_code << 16);                       // wg_mask=0, data_size
  g1[1] = (int)((elems & 0xffffu) << 16);                    // tensor_dim0[15:0] @ bits63:48
  g1[2] = (int)(((elems >> 16) & 0xffffu) | (1u << 16));     // tensor_dim0[31:16] | tensor_dim1=1
  g1[3] = (int)((elems & 0xffffu) << 16);                    // tile_dim0 @ bits127:112
  g1[4] = 0;                                                 // tile_dim1=0 (unused), tile_dim2=0
  g1[5] = (int)elems;                                        // tensor_dim0_stride[31:0]
  g1[6] = 0;
  g1[7] = 0;
  i32x4 z4; z4[0] = 0; z4[1] = 0; z4[2] = 0; z4[3] = 0;      // groups 2/3 unused (<=2D)
  i32x8 z8; z8[0] = 0; z8[1] = 0; z8[2] = 0; z8[3] = 0;
  z8[4] = 0; z8[5] = 0; z8[6] = 0; z8[7] = 0;
  __builtin_amdgcn_tensor_load_to_lds(g0, g1, z4, z4, z8, 0);
}

// ---- grid barrier across the persistent scan WGs ---------------------------
__device__ __forceinline__ void gru5_grid_barrier(unsigned* bar, unsigned nwg, unsigned target) {
  __syncthreads();
  __builtin_amdgcn_s_cluster_barrier();   // HW fast path when clustered; NOP otherwise
  if (threadIdx.x == 0) {
    __hip_atomic_fetch_add(bar, 1u, __ATOMIC_RELEASE, __HIP_MEMORY_SCOPE_AGENT);
    while (__hip_atomic_load(bar, __ATOMIC_ACQUIRE, __HIP_MEMORY_SCOPE_AGENT) < nwg * target)
      __builtin_amdgcn_s_sleep(8);
  }
  __syncthreads();
}

// ---- kernel: embedding gather, fp32 -> f16, layout [s*B+b][DI] -------------
__global__ __launch_bounds__(256) void gru5_embed(const int* __restrict__ tok,
                                                  const float* __restrict__ emb,
                                                  _Float16* __restrict__ xe) {
  using namespace gru5;
  const int row = blockIdx.x;           // s*B + b
  const int s = row >> 5, b = row & 31;
  const int t = tok[b * S + s];
  xe[(size_t)row * DI + threadIdx.x] = (_Float16)emb[(size_t)t * DI + threadIdx.x];
}

// ---- kernel: cast+transpose weights: out[n*K + k] = W[(r0+k)*ldw + n] ------
__global__ __launch_bounds__(256) void gru5_castT(const float* __restrict__ W, int ldw,
                                                  int r0, int K,
                                                  _Float16* __restrict__ out) {
  const int idx = blockIdx.x * 256 + threadIdx.x; // n*K + k
  const int n = idx / K, k = idx - n * K;
  out[idx] = (_Float16)W[(size_t)(r0 + k) * ldw + n];
}

// ---- kernel: WMMA GEMM  C[M x N] = A[M x K] * Bt[N x K]^T ------------------
// one 16x32 C super-tile per wave (A fragment reused across two B tiles).
__global__ __launch_bounds__(256) void gru5_gemm(const _Float16* __restrict__ A, int lda,
                                                 const _Float16* __restrict__ Bt, int ldb,
                                                 _Float16* __restrict__ C, int ldc,
                                                 int Mr, int Nc, int K) {
  const int lane = threadIdx.x & 31;
  const int wid  = threadIdx.x >> 5;
  const int tn2  = Nc >> 5;
  const int total = (Mr >> 4) * tn2;
  const int tile = blockIdx.x * 8 + wid;
  if (tile >= total) return;
  const int m0 = (tile / tn2) << 4;
  const int n0 = (tile - (tile / tn2) * tn2) << 5;
  f8v c0 = {}, c1 = {};
  for (int k0 = 0; k0 < K; k0 += 32) {
    __builtin_prefetch(A + (size_t)(m0 + (lane & 15)) * lda + k0 + 64, 0, 1);
    h16v a  = gru5_afrag(A, lda, m0, k0, lane);
    h16v b0 = gru5_bfrag(Bt, ldb, n0,      k0, lane);
    h16v b1 = gru5_bfrag(Bt, ldb, n0 + 16, k0, lane);
    c0 = gru5_wmma(a, b0, c0);
    c1 = gru5_wmma(a, b1, c1);
  }
  const int colc = n0 + (lane & 15);
  const int rb   = m0 + ((lane >> 4) << 3);
#pragma unroll
  for (int r = 0; r < 8; ++r) {
    C[(size_t)(rb + r) * ldc + colc]      = (_Float16)c0[r];
    C[(size_t)(rb + r) * ldc + colc + 16] = (_Float16)c1[r];
  }
}

// ---- kernel: persistent GRU scan for one layer -----------------------------
// 8 WGs x 256 thr (64 waves). Phase1: 64 waves = 64 (z|r) column tiles of 16,
// both M-tiles each. Phase2: waves 0..31 = n-gate tiles + elementwise combine.
// Recurrent weight slices + H mirror staged by TDM into 288KB of WGP LDS.
__global__ __launch_bounds__(256) void gru5_scan(const _Float16* __restrict__ gx,   // [S*B][1536]
                                                 const _Float16* __restrict__ WhT,  // [3][DH][DH] as [n][k]
                                                 const float* __restrict__ bz,
                                                 const float* __restrict__ br,
                                                 const float* __restrict__ bn,
                                                 _Float16* __restrict__ Hg,         // [B][DH] hidden (zeroed)
                                                 _Float16* __restrict__ Rh,         // [B][DH] r*H scratch
                                                 float*    __restrict__ Zb,         // [B][DH] z scratch
                                                 _Float16* __restrict__ hseq,       // [S*B][DH] out
                                                 unsigned* __restrict__ bar) {
  using namespace gru5;
  extern __shared__ _Float16 smem[];
  _Float16* slab1 = smem;                   // [8 waves][16 cols][DH]   (z|r slices)
  _Float16* slab2 = smem + 8 * 16 * DH;     // [8 waves][16 cols][DH]   (n slices)
  _Float16* Hl    = smem + 2 * 8 * 16 * DH; // [B][DH]

  const int tid  = threadIdx.x;
  const int lane = tid & 31;
  const int wid  = tid >> 5;
  const int gw   = blockIdx.x * 8 + wid;    // 0..63

  // ---- Stage weight slabs via TDM: one contiguous 128KB DMA per slab. ------
  // Block b waves own contiguous column tiles, so the per-wave [16 col][DH]
  // slices concatenate to 128 contiguous [n][k] rows of WhT.
  if (wid == 0) {
    const unsigned g1   = (blockIdx.x >> 2) & 1;          // blocks 0-3: z, 4-7: r
    const unsigned row0 = (blockIdx.x & 3) * 128;
    gru5_tdm_load_1d(WhT + ((size_t)g1 * DH + row0) * DH, 0u, SLAB_BYTES / 8, 3);
    if (blockIdx.x < 4)
      gru5_tdm_load_1d(WhT + ((size_t)2 * DH + row0) * DH, SLAB_BYTES, SLAB_BYTES / 8, 3);
    __builtin_amdgcn_s_wait_tensorcnt(0);
  }
  __syncthreads();

  const int g   = gw >> 5;
  const int col = ((gw & 31) << 4) + (lane & 15);
  const float* bias1 = g ? br : bz;
  const _Float16* my1 = slab1 + wid * 16 * DH;
  const _Float16* my2 = slab2 + wid * 16 * DH;

  unsigned bt = 0;
  for (int t = 0; t < S; ++t) {
    // Mirror H into LDS with one 32KB TDM DMA.
    if (wid == 0) {
      gru5_tdm_load_1d(Hg, 2u * SLAB_BYTES, (unsigned)(B * DH * sizeof(_Float16)) / 8, 3);
      __builtin_amdgcn_s_wait_tensorcnt(0);
    }
    __syncthreads();

    // ---- phase 1: z and r gates ----
#pragma unroll
    for (int mt = 0; mt < 2; ++mt) {
      const int rb = mt * 16 + ((lane >> 4) << 3);
      f8v c;
#pragma unroll
      for (int r = 0; r < 8; ++r)
        c[r] = (float)gx[((size_t)t * B + rb + r) * G3 + g * DH + col] + bias1[col];
#pragma unroll 4
      for (int k0 = 0; k0 < DH; k0 += 32) {
        h16v a  = gru5_afrag(Hl, DH, mt * 16, k0, lane);
        h16v bm = gru5_bfrag(my1, DH, 0, k0, lane);
        c = gru5_wmma(a, bm, c);
      }
#pragma unroll
      for (int r = 0; r < 8; ++r) {
        const int b = rb + r;
        const float v = gru5_sigmoid(c[r]);
        if (g == 0) Zb[b * DH + col] = v;
        else        Rh[b * DH + col] = (_Float16)(v * (float)Hl[b * DH + col]);
      }
    }
    __threadfence();
    gru5_grid_barrier(bar, NWG, ++bt);

    // ---- phase 2: candidate gate + combine (waves 0..31) ----
    if (gw < 32) {
#pragma unroll
      for (int mt = 0; mt < 2; ++mt) {
        const int rb = mt * 16 + ((lane >> 4) << 3);
        f8v c;
#pragma unroll
        for (int r = 0; r < 8; ++r)
          c[r] = (float)gx[((size_t)t * B + rb + r) * G3 + 2 * DH + col] + bn[col];
#pragma unroll 4
        for (int k0 = 0; k0 < DH; k0 += 32) {
          h16v a  = gru5_afrag(Rh, DH, mt * 16, k0, lane);  // (r .* H) from L2
          h16v bm = gru5_bfrag(my2, DH, 0, k0, lane);
          c = gru5_wmma(a, bm, c);
        }
#pragma unroll
        for (int r = 0; r < 8; ++r) {
          const int b = rb + r;
          const float ht = tanhf(c[r]);
          const float z  = Zb[b * DH + col];
          const float h0 = (float)Hl[b * DH + col];
          const float hn = (1.0f - z) * h0 + z * ht;
          Hg[b * DH + col] = (_Float16)hn;
          hseq[((size_t)t * B + b) * DH + col] = (_Float16)hn;
        }
      }
    }
    __threadfence();
    gru5_grid_barrier(bar, NWG, ++bt);
  }
}

// ---- kernel: FC bias + log_softmax over DO=128, one wave per row -----------
__global__ __launch_bounds__(256) void gru5_logsoftmax(const _Float16* __restrict__ lt,
                                                       const float* __restrict__ bias,
                                                       float* __restrict__ out) {
  using namespace gru5;
  const int lane = threadIdx.x & 31;
  const int wid  = threadIdx.x >> 5;
  const int row  = blockIdx.x * 8 + wid;   // s*B + b
  float v[4];
  float mx = -1e30f;
#pragma unroll
  for (int i = 0; i < 4; ++i) {
    const int c = lane + i * 32;
    v[i] = (float)lt[(size_t)row * DO + c] + bias[c];
    mx = fmaxf(mx, v[i]);
  }
#pragma unroll
  for (int o = 16; o > 0; o >>= 1) mx = fmaxf(mx, __shfl_xor(mx, o, 32));
  float sum = 0.f;
#pragma unroll
  for (int i = 0; i < 4; ++i) sum += __expf(v[i] - mx);
#pragma unroll
  for (int o = 16; o > 0; o >>= 1) sum += __shfl_xor(sum, o, 32);
  const float lse = mx + __logf(sum);
  const int s = row >> 5, b = row & 31;
  float* orow = out + ((size_t)b * S + s) * DO;
#pragma unroll
  for (int i = 0; i < 4; ++i) orow[lane + i * 32] = v[i] - lse;
}

// ---------------------------------------------------------------------------
extern "C" void kernel_launch(void* const* d_in, const int* in_sizes, int n_in,
                              void* d_out, int out_size, void* d_ws, size_t ws_size,
                              hipStream_t stream) {
  using namespace gru5;
  (void)in_sizes; (void)n_in; (void)out_size; (void)ws_size;

  const int*   tokens = (const int*)d_in[0];
  const float* emb    = (const float*)d_in[1];
  const float* l0W[3] = { (const float*)d_in[2], (const float*)d_in[4], (const float*)d_in[6] };
  const float* l0b[3] = { (const float*)d_in[3], (const float*)d_in[5], (const float*)d_in[7] };
  const float* l1W[3] = { (const float*)d_in[8], (const float*)d_in[10], (const float*)d_in[12] };
  const float* l1b[3] = { (const float*)d_in[9], (const float*)d_in[11], (const float*)d_in[13] };
  const float* fcW    = (const float*)d_in[14];
  const float* fcb    = (const float*)d_in[15];
  float* out = (float*)d_out;

  // ---- workspace carve-up (bytes) ----
  char* ws = (char*)d_ws;
  size_t off = 0;
  auto take = [&](size_t bytes) { char* p = ws + off; off += (bytes + 255) & ~size_t(255); return p; };
  unsigned*  bar   = (unsigned*) take(256);
  _Float16*  xe    = (_Float16*) take((size_t)M * DI * 2);   //  8 MB
  _Float16*  h0seq = (_Float16*) take((size_t)M * DH * 2);   // 16 MB
  _Float16*  h1seq = (_Float16*) take((size_t)M * DH * 2);   // 16 MB
  _Float16*  gx    = (_Float16*) take((size_t)M * G3 * 2);   // 48 MB (reused by both layers)
  _Float16*  W0xT  = (_Float16*) take((size_t)3 * DH * DI * 2);
  _Float16*  W0hT  = (_Float16*) take((size_t)3 * DH * DH * 2);
  _Float16*  W1xT  = (_Float16*) take((size_t)3 * DH * DH * 2);
  _Float16*  W1hT  = (_Float16*) take((size_t)3 * DH * DH * 2);
  _Float16*  WfcT  = (_Float16*) take((size_t)DO * DH * 2);
  _Float16*  Hg    = (_Float16*) take((size_t)B * DH * 2);
  _Float16*  Rh    = (_Float16*) take((size_t)B * DH * 2);
  float*     Zb    = (float*)    take((size_t)B * DH * 4);
  _Float16*  ltmp  = (_Float16*) take((size_t)M * DO * 2);   //  4 MB

  constexpr unsigned LDSB = (2u * 8 * 16 * DH + B * DH) * sizeof(_Float16); // 288 KB

  // ---- weight prep (f16, [N][K] transposed images) ----
  for (int g = 0; g < 3; ++g) {
    gru5_castT<<<(DH * DI) / 256, 256, 0, stream>>>(l0W[g], DH, 0,   DI, W0xT + (size_t)g * DH * DI);
    gru5_castT<<<(DH * DH) / 256, 256, 0, stream>>>(l0W[g], DH, DI,  DH, W0hT + (size_t)g * DH * DH);
    gru5_castT<<<(DH * DH) / 256, 256, 0, stream>>>(l1W[g], DH, 0,   DH, W1xT + (size_t)g * DH * DH);
    gru5_castT<<<(DH * DH) / 256, 256, 0, stream>>>(l1W[g], DH, DH,  DH, W1hT + (size_t)g * DH * DH);
  }
  gru5_castT<<<(DO * DH) / 256, 256, 0, stream>>>(fcW, DO, 0, DH, WfcT);

  // ---- embedding gather ----
  gru5_embed<<<M, 256, 0, stream>>>(tokens, emb, xe);

  // ---- layer 0: parallel x-projections, then sequential scan ----
  gru5_gemm<<<(M / 16) * (G3 / 32) / 8, 256, 0, stream>>>(xe, DI, W0xT, DI, gx, G3, M, G3, DI);
  (void)hipMemsetAsync(Hg, 0, (size_t)B * DH * 2, stream);
  (void)hipMemsetAsync(bar, 0, 256, stream);
  gru5_scan<<<NWG, 256, LDSB, stream>>>(gx, W0hT, l0b[0], l0b[1], l0b[2], Hg, Rh, Zb, h0seq, bar);

  // ---- layer 1 ----
  gru5_gemm<<<(M / 16) * (G3 / 32) / 8, 256, 0, stream>>>(h0seq, DH, W1xT, DH, gx, G3, M, G3, DH);
  (void)hipMemsetAsync(Hg, 0, (size_t)B * DH * 2, stream);
  (void)hipMemsetAsync(bar, 0, 256, stream);
  gru5_scan<<<NWG, 256, LDSB, stream>>>(gx, W1hT, l1b[0], l1b[1], l1b[2], Hg, Rh, Zb, h1seq, bar);

  // ---- FC + log_softmax ----
  gru5_gemm<<<(M / 16) * (DO / 32) / 8, 256, 0, stream>>>(h1seq, DH, WfcT, DH, ltmp, DO, M, DO, DH);
  gru5_logsoftmax<<<M / 8, 256, 0, stream>>>(ltmp, fcb, out);
}
